// BandSplitModule_10660108829566
// MI455X (gfx1250) — matrix operand order repaired and checked by reading, not compiled, not run on hardware
//
#include <hip/hip_runtime.h>
#include <hip/hip_bf16.h>

typedef __attribute__((ext_vector_type(16))) _Float16 v16h;
typedef __attribute__((ext_vector_type(8)))  _Float16 v8h;
typedef __attribute__((ext_vector_type(8)))  float    v8f;

constexpr int TT    = 256;   // time tile per workgroup
constexpr int NF    = 128;   // features
constexpr int KTOT  = 32;    // total bands in output
constexpr int TDIM  = 2048;  // T
constexpr int FBINS = 1025;  // F

// One workgroup: one (batch, band, 256-wide t tile).
// 256 threads = 8 waves. Wave w computes features [16w, 16w+16) x all 256 t.
template <int WB>
__global__ __launch_bounds__(256, 1)
void bandsplit_wmma(const float* __restrict__ xr, const float* __restrict__ xi,
                    const float* __restrict__ gg, const float* __restrict__ bb,
                    const float* __restrict__ Wm, const float* __restrict__ cb,
                    float* __restrict__ out, int s_bin, int kb_off)
{
    constexpr int C  = 2 * WB;   // channels per band
    constexpr int CP = C + 8;    // padded f16 row pitch (byte pitch % 16 == 0)

    __shared__ _Float16 Ax[TT][CP];  // normalized X: [t][ch]  (WMMA B operand)
    __shared__ _Float16 Wt[NF][CP];  // W transposed: [feat][ch] (WMMA A operand)

    const int tid  = threadIdx.x;
    const int b    = blockIdx.z;
    const int jb   = blockIdx.y;          // band index within group
    const int kb   = kb_off + jb;         // global band index
    const int t0   = blockIdx.x * TT;
    const int bin0 = s_bin + jb * WB;

    // ---- stage W^T into LDS as f16: Wt[n][ch] = W[jb][ch][n] ----
    {
        const float* Wg = Wm + (size_t)jb * C * NF;
        for (int idx = tid; idx < C * NF; idx += 256) {  // idx = ch*NF + n, coalesced reads
            int ch = idx >> 7;        // / NF
            int n  = idx & (NF - 1);
            Wt[n][ch] = (_Float16)Wg[idx];
        }
    }

    // ---- phase 1: per-t channel LayerNorm + affine, f16 into Ax[t][ch] ----
    {
        const size_t tg  = (size_t)t0 + tid;
        const float* xrp = xr + ((size_t)b * FBINS + bin0) * TDIM + tg;
        const float* xip = xi + ((size_t)b * FBINS + bin0) * TDIM + tg;
        float vals[C];
        float sum = 0.f, sq = 0.f;
#pragma unroll
        for (int j = 0; j < WB; ++j) {
            float r  = xrp[(size_t)j * TDIM];   // coalesced over t across lanes
            float im = xip[(size_t)j * TDIM];
            vals[j]      = r;
            vals[WB + j] = im;
            sum += r + im;
            sq  += r * r + im * im;
        }
        const float inv  = 1.0f / (float)C;
        const float mu   = sum * inv;
        const float var  = sq * inv - mu * mu;   // population variance
        const float rstd = rsqrtf(var + 1e-8f);
        const float* gp  = gg + jb * C;
        const float* bp  = bb + jb * C;
#pragma unroll
        for (int ch = 0; ch < C; ++ch) {
            Ax[tid][ch] = (_Float16)((vals[ch] - mu) * rstd * gp[ch] + bp[ch]);
        }
    }
    __syncthreads();

    // ---- phase 2: D[feat][t] = sum_ch Wt[feat][ch] * Ax[t][ch] via WMMA ----
    const int lane = tid & 31;
    const int ln   = lane & 15;   // A: M index / B,C,D: N (=t) index
    const int h    = lane >> 4;   // lane half
    const int m0   = (tid >> 5) * 16;  // feature subtile base for this wave

    v8f acc[TT / 16] = {};        // 16 accumulators (one per 16-t column block)

#pragma unroll
    for (int ks = 0; ks < C; ks += 32) {
        // A fragment (16-bit A 16x32 layout): half h holds K {8h..8h+7, 16+8h..16+8h+7}
        v16h a;
        {
            const v8h alo = *(const v8h*)&Wt[m0 + ln][ks + 8 * h];
            const v8h ahi = *(const v8h*)&Wt[m0 + ln][ks + 16 + 8 * h];
#pragma unroll
            for (int e = 0; e < 8; ++e) { a[e] = alo[e]; a[8 + e] = ahi[e]; }
        }
#pragma unroll
        for (int ns = 0; ns < TT / 16; ++ns) {
            // B fragment (K x N): lane half h holds K {ks+8h..ks+8h+7, ks+16+8h..ks+16+8h+7}, column t = ns*16+ln
            const v8h blo = *(const v8h*)&Ax[ns * 16 + ln][ks + 8 * h];
            const v8h bhi = *(const v8h*)&Ax[ns * 16 + ln][ks + 16 + 8 * h];
            v16h bf;
#pragma unroll
            for (int e = 0; e < 8; ++e) { bf[e] = blo[e]; bf[8 + e] = bhi[e]; }
            acc[ns] = __builtin_amdgcn_wmma_f32_16x16x32_f16(
                false, a, false, bf, (short)0, acc[ns], false, false);
        }
    }

    // ---- epilogue: add bias, t-contiguous stores into out[b][feat][kb][t] ----
    // C/D layout: VGPR v, lane half h -> M(feat) = m0 + v + 8h ; N(t) = ns*16 + ln
    const float* cbp = cb + jb * NF + m0 + 8 * h;
    const size_t obase =
        (((size_t)b * NF + (size_t)(m0 + 8 * h)) * KTOT + kb) * TDIM + t0 + ln;
#pragma unroll
    for (int v = 0; v < 8; ++v) {
        const float  bias = cbp[v];
        const size_t orow = obase + (size_t)v * (size_t)KTOT * TDIM;
#pragma unroll
        for (int ns = 0; ns < TT / 16; ++ns) {
            out[orow + ns * 16] = acc[ns][v] + bias;  // 16 lanes -> 64B contiguous
        }
    }
}

extern "C" void kernel_launch(void* const* d_in, const int* in_sizes, int n_in,
                              void* d_out, int out_size, void* d_ws, size_t ws_size,
                              hipStream_t stream)
{
    const float* xr  = (const float*)d_in[0];
    const float* xi  = (const float*)d_in[1];
    // d_in[2] = sample_rate (== n_fft == 2048 in this problem; bins == Hz)
    const float* g16 = (const float*)d_in[3];
    const float* b16 = (const float*)d_in[4];
    const float* W16 = (const float*)d_in[5];
    const float* c16 = (const float*)d_in[6];
    const float* g32 = (const float*)d_in[7];
    const float* b32 = (const float*)d_in[8];
    const float* W32 = (const float*)d_in[9];
    const float* c32 = (const float*)d_in[10];
    const float* g64 = (const float*)d_in[11];
    const float* b64 = (const float*)d_in[12];
    const float* W64 = (const float*)d_in[13];
    const float* c64 = (const float*)d_in[14];
    float* out = (float*)d_out;

    const int B = in_sizes[0] / (FBINS * TDIM);
    dim3 blk(256, 1, 1);

    // group16: bands 0..15, bins [0,256), width 16, C=32
    hipLaunchKernelGGL((bandsplit_wmma<16>), dim3(TDIM / TT, 16, B), blk, 0, stream,
                       xr, xi, g16, b16, W16, c16, out, 0, 0);
    // group32: bands 16..23, bins [256,512), width 32, C=64
    hipLaunchKernelGGL((bandsplit_wmma<32>), dim3(TDIM / TT, 8, B), blk, 0, stream,
                       xr, xi, g32, b32, W32, c32, out, 256, 16);
    // group64: bands 24..31, bins [512,1024), width 64, C=128
    hipLaunchKernelGGL((bandsplit_wmma<64>), dim3(TDIM / TT, 8, B), blk, 0, stream,
                       xr, xi, g64, b64, W64, c64, out, 512, 24);
}
